// AttentionBlock_75617194213468
// MI455X (gfx1250) — compile-verified, matrix operand order checked by reading, not run on hardware
//
#include <hip/hip_runtime.h>
#include <hip/hip_bf16.h>

// ---------------- problem constants ----------------
#define BATCH   16
#define CCH     512        // C
#define LTOK    1024       // H*W
#define NHEADS  8
#define HEADD   64
#define NGROUPS 8
#define LCOND   77
#define CONDC   768
#define LCP     128        // cond rows padded (multiple of 64)
#define LKREAL  1101       // L + Lc
#define LKP     1120       // padded keys (multiple of 32)

typedef __attribute__((ext_vector_type(16))) _Float16 v16h;
typedef __attribute__((ext_vector_type(8)))  _Float16 v8h;
typedef __attribute__((ext_vector_type(8)))  float    v8f;
typedef __attribute__((ext_vector_type(4)))  int      v4i;

union frag16 { v16h v; v8h h[2]; };

static __device__ inline v8f wmma_f16(v16h a, v16h b, v8f c) {
    return __builtin_amdgcn_wmma_f32_16x16x32_f16(false, a, false, b, (short)0, c,
                                                  false, false);
}

// ---- CDNA5 async global->LDS copy (ASYNCcnt path) ----
// Round-3 probe proved __builtin_amdgcn_global_load_async_to_lds_b128 exists
// (signature: (v4i AS1*, v4i AS3*, imm offset, imm cpol)). Use it whenever
// declared; the counter wait falls back to inline asm if its builtin is absent.
#if __has_builtin(__builtin_amdgcn_global_load_async_to_lds_b128)
#define HAVE_ASYNC_LDS 1
#else
#define HAVE_ASYNC_LDS 0
#endif

static __device__ inline void async_copy16(_Float16* dst, const _Float16* src) {
#if HAVE_ASYNC_LDS
    __builtin_amdgcn_global_load_async_to_lds_b128(
        (__attribute__((address_space(1))) v4i*)src,
        (__attribute__((address_space(3))) v4i*)dst, 0, 0);
#else
    *(v8h*)dst = *(const v8h*)src;
#endif
}
template <int N>
static __device__ inline void wait_async() {
#if HAVE_ASYNC_LDS
#if __has_builtin(__builtin_amdgcn_s_wait_asynccnt)
    __builtin_amdgcn_s_wait_asynccnt(N);
#else
    if constexpr (N == 0) {
        asm volatile("s_wait_asynccnt 0x0" ::: "memory");
    } else {
        asm volatile("s_wait_asynccnt 0x4" ::: "memory");
    }
#endif
#endif
}

// ---- DPP16 butterfly reduction within a 16-lane row (pure VALU, no LDS) ----
template <int CTRL>
static __device__ inline float fmax_dpp(float x) {
    int v = __builtin_amdgcn_mov_dpp(__builtin_bit_cast(int, x), CTRL, 0xF, 0xF, true);
    return fmaxf(x, __builtin_bit_cast(float, v));
}
template <int CTRL>
static __device__ inline float fadd_dpp(float x) {
    int v = __builtin_amdgcn_mov_dpp(__builtin_bit_cast(int, x), CTRL, 0xF, 0xF, true);
    return x + __builtin_bit_cast(float, v);
}
static __device__ inline float rowmax16(float x) {
    x = fmax_dpp<0xB1>(x);   // quad_perm xor1
    x = fmax_dpp<0x4E>(x);   // quad_perm xor2
    x = fmax_dpp<0x141>(x);  // row_half_mirror
    x = fmax_dpp<0x140>(x);  // row_mirror
    return x;
}
static __device__ inline float rowsum16(float x) {
    x = fadd_dpp<0xB1>(x);
    x = fadd_dpp<0x4E>(x);
    x = fadd_dpp<0x141>(x);
    x = fadd_dpp<0x140>(x);
    return x;
}

// ---------------- GroupNorm: stats per (b,g) ----------------
__global__ void __launch_bounds__(256) gn_stats_kernel(const float* __restrict__ x0,
                                                       float* __restrict__ stats) {
    __shared__ float s1[256], s2[256];
    const int bg = blockIdx.x;                      // b*NGROUPS + g
    const float* p = x0 + (long)bg * 65536;         // group = 64 ch * 1024 contiguous
    float a = 0.f, b = 0.f;
    for (int i = threadIdx.x; i < 65536; i += 256) {
        float v = p[i];
        a += v; b += v * v;
    }
    s1[threadIdx.x] = a; s2[threadIdx.x] = b;
    __syncthreads();
    for (int s = 128; s > 0; s >>= 1) {
        if (threadIdx.x < s) {
            s1[threadIdx.x] += s1[threadIdx.x + s];
            s2[threadIdx.x] += s2[threadIdx.x + s];
        }
        __syncthreads();
    }
    if (threadIdx.x == 0) {
        float mu  = s1[0] * (1.f / 65536.f);
        float var = s2[0] * (1.f / 65536.f) - mu * mu;
        stats[bg * 2]     = mu;
        stats[bg * 2 + 1] = rsqrtf(var + 1e-5f);
    }
}

// ---------------- GroupNorm apply + transpose to tokens (fp16) ----------------
__global__ void __launch_bounds__(256) gn_apply_kernel(const float* __restrict__ x0,
                                                       const float* __restrict__ stats,
                                                       const float* __restrict__ gamma,
                                                       const float* __restrict__ beta,
                                                       _Float16* __restrict__ x16) {
    const int bc = blockIdx.x;          // b*C + c
    const int c  = bc & (CCH - 1);
    const int b  = bc >> 9;
    const int g  = c >> 6;
    const float mu = stats[(b * NGROUPS + g) * 2];
    const float rs = stats[(b * NGROUPS + g) * 2 + 1];
    const float ga = gamma[c] * rs;
    const float be = beta[c] - mu * ga;
    const float* src = x0 + (long)bc * LTOK;
    _Float16* dst = x16 + (long)b * LTOK * CCH + c;
    for (int l = threadIdx.x; l < LTOK; l += 256)
        dst[(long)l * CCH] = (_Float16)(src[l] * ga + be);
}

// ---------------- f32 -> f16 convert ----------------
__global__ void __launch_bounds__(256) cvt_f16_kernel(const float* __restrict__ s,
                                                      _Float16* __restrict__ d, int n) {
    int i = blockIdx.x * 256 + threadIdx.x;
    if (i < n) d[i] = (_Float16)s[i];
}

// ---------------- cond -> fp16, padded rows ----------------
__global__ void __launch_bounds__(256) pad_cond_kernel(const float* __restrict__ cond,
                                                       _Float16* __restrict__ c16) {
    long i = (long)blockIdx.x * 256 + threadIdx.x;    // over B*LCP*CONDC
    if (i >= (long)BATCH * LCP * CONDC) return;
    int c = (int)(i % CONDC);
    long t = i / CONDC;
    int r = (int)(t % LCP);
    int b = (int)(t / LCP);
    float v = (r < LCOND) ? cond[((long)(b * LCOND + r)) * CONDC + c] : 0.f;
    c16[i] = (_Float16)v;
}

// ---------------- WMMA GEMM: D = A[M,K] * B[N,K]^T (+bias) ----------------
// MODE 0: fp16 output, row-major [M, ldd]
// MODE 1: fp32 output transposed [N, M] with residual add (out projection)
#define BM 64
#define BN 64
#define BK 32
#define LDT 40

template <int MODE>
__global__ void __launch_bounds__(128) gemm_kernel(
    const _Float16* __restrict__ A, long strideAb, int lda,
    const _Float16* __restrict__ Bm, int ldb,
    const float* __restrict__ bias,
    _Float16* __restrict__ Dh, long strideDb, int ldd,
    float* __restrict__ Df, const float* __restrict__ X0, long strideFb,
    int M, int N, int K) {
    __shared__ _Float16 As[2][BM][LDT];
    __shared__ _Float16 Bs[2][BN][LDT];
    const int tilesN = N / BN;
    const int tm = (blockIdx.x / tilesN) * BM;
    const int tn = (blockIdx.x % tilesN) * BN;
    const int b  = blockIdx.y;
    const _Float16* Ab = A + (long)b * strideAb;
    const int tid = threadIdx.x;
    const int lane = tid & 31, wave = tid >> 5;
    const int wm = (wave & 1) * 32, wn = (wave >> 1) * 32;
    const int fr = lane & 15, kb = (lane >> 4) * 8;

    // stage one k-tile (4 async 16B copies per thread)
    auto stage = [&](int buf, int kk) {
#pragma unroll
        for (int it = 0; it < 2; ++it) {
            int s = tid + it * 128;
            int r = s >> 2, c8 = (s & 3) * 8;
            async_copy16(&As[buf][r][c8], &Ab[(long)(tm + r) * lda + kk + c8]);
            async_copy16(&Bs[buf][r][c8], &Bm[(long)(tn + r) * ldb + kk + c8]);
        }
    };

    v8f acc[2][2] = {};
    stage(0, 0);
    int buf = 0;
    for (int kk = 0; kk < K; kk += BK) {
        const bool more = (kk + BK) < K;
        if (more) stage(buf ^ 1, kk + BK);          // prefetch next tile into other buffer
        if (more) wait_async<4>(); else wait_async<0>();  // drain current tile's copies
        __syncthreads();
        frag16 af[2], bf[2];
#pragma unroll
        for (int i = 0; i < 2; i++) {
            af[i].h[0] = *(const v8h*)&As[buf][wm + i * 16 + fr][kb];
            af[i].h[1] = *(const v8h*)&As[buf][wm + i * 16 + fr][kb + 16];
            bf[i].h[0] = *(const v8h*)&Bs[buf][wn + i * 16 + fr][kb];
            bf[i].h[1] = *(const v8h*)&Bs[buf][wn + i * 16 + fr][kb + 16];
        }
#pragma unroll
        for (int i = 0; i < 2; i++)
#pragma unroll
            for (int j = 0; j < 2; j++)
                acc[i][j] = wmma_f16(af[i].v, bf[j].v, acc[i][j]);
        __syncthreads();
        buf ^= 1;
    }
    const int rbase = (lane >> 4) * 8;
    const int col = lane & 15;
#pragma unroll
    for (int i = 0; i < 2; i++)
#pragma unroll
        for (int j = 0; j < 2; j++) {
            int n = tn + wn + j * 16 + col;
            float bv = bias[n];
#pragma unroll
            for (int e = 0; e < 8; e++) {
                int m = tm + wm + i * 16 + rbase + e;
                float v = acc[i][j][e] + bv;
                if (MODE == 0) {
                    Dh[(long)b * strideDb + (long)m * ldd + n] = (_Float16)v;
                } else {
                    long off = (long)b * strideFb + (long)n * M + m;
                    Df[off] = v + X0[off];
                }
            }
        }
}

// ---------------- repack K (row-major) and V (transposed), padded ----------------
__global__ void __launch_bounds__(256) build_kv_kernel(const _Float16* __restrict__ qkv,
                                                       const _Float16* __restrict__ kvc,
                                                       _Float16* __restrict__ Kh,
                                                       _Float16* __restrict__ Vt) {
    long idx = (long)blockIdx.x * 256 + threadIdx.x;   // B*NH*LKP*HEADD
    if (idx >= (long)BATCH * NHEADS * LKP * HEADD) return;
    int d = (int)(idx % HEADD);
    long t = idx / HEADD;
    int key = (int)(t % LKP); t /= LKP;
    int h = (int)(t % NHEADS);
    int b = (int)(t / NHEADS);
    _Float16 kv = (_Float16)0.f, vv = (_Float16)0.f;
    if (key < LTOK) {
        const _Float16* base = qkv + ((long)(b * LTOK + key)) * (3 * CCH);
        kv = base[CCH + h * HEADD + d];
        vv = base[2 * CCH + h * HEADD + d];
    } else if (key < LKREAL) {
        const _Float16* base = kvc + ((long)(b * LCP + (key - LTOK))) * (2 * CCH);
        kv = base[h * HEADD + d];
        vv = base[CCH + h * HEADD + d];
    }
    Kh[((long)(b * NHEADS + h) * LKP + key) * HEADD + d] = kv;
    Vt[((long)(b * NHEADS + h) * HEADD + d) * LKP + key] = vv;
}

// ---------------- flash attention ----------------
// Block = 4 waves, one (b, h, 64-query tile). Double-buffered K/V staging in
// LDS shared by all waves; each wave owns 16 query rows.
__global__ void __launch_bounds__(128) attn_kernel(const _Float16* __restrict__ qkv,
                                                   const _Float16* __restrict__ Kh,
                                                   const _Float16* __restrict__ Vt,
                                                   _Float16* __restrict__ O) {
    __shared__ _Float16 Ks[2][32][72];     // 32 keys x 64 d   (+8 pad)
    __shared__ _Float16 Vs[2][64][LDT];    // 64 d   x 32 keys (+8 pad)
    __shared__ _Float16 Pt[4][16][LDT];    // per-wave P tile
    const int tid = threadIdx.x;
    const int lane = tid & 31, wave = tid >> 5;
    const int qt = blockIdx.x & 15;                 // 16 q-tiles of 64
    const int h  = (blockIdx.x >> 4) & 7;
    const int b  = blockIdx.x >> 7;
    const int q0 = qt * 64 + wave * 16;
    const int fr = lane & 15, kb = (lane >> 4) * 8;

    // Q fragments (resident for the whole loop)
    const _Float16* qrow = qkv + ((long)(b * LTOK + q0 + fr)) * (3 * CCH) + h * HEADD;
    frag16 qf[2];
#pragma unroll
    for (int c = 0; c < 2; c++) {
        qf[c].h[0] = *(const v8h*)&qrow[c * 32 + kb];
        qf[c].h[1] = *(const v8h*)&qrow[c * 32 + kb + 16];
    }
    const _Float16* Kb = Kh + (long)(b * NHEADS + h) * LKP * HEADD;
    const _Float16* Vb = Vt + (long)(b * NHEADS + h) * HEADD * LKP;

    // stage K (32x64) + V^T (64x32) for one k-step (4 async copies / thread)
    auto stage = [&](int buf, int kk) {
#pragma unroll
        for (int it = 0; it < 2; ++it) {
            int s = tid + it * 128;
            int rk = s >> 3, ck = (s & 7) * 8;
            async_copy16(&Ks[buf][rk][ck], Kb + (long)(kk + rk) * HEADD + ck);
            int rv = s >> 2, cv = (s & 3) * 8;
            async_copy16(&Vs[buf][rv][cv], Vb + (long)rv * LKP + kk + cv);
        }
    };

    float mrow[8], lrow[8];
    v8f o[4] = {};
#pragma unroll
    for (int i = 0; i < 8; i++) { mrow[i] = -1e30f; lrow[i] = 0.f; }
    const float SC = 0.125f * 1.44269504088896340736f;   // scale * log2(e)

    stage(0, 0);
    int buf = 0;
    for (int kk = 0; kk < LKP; kk += 32) {
        const bool more = (kk + 32) < LKP;
        if (more) stage(buf ^ 1, kk + 32);
        if (more) wait_async<4>(); else wait_async<0>();
        __syncthreads();

        // ---- scores: Q[16x64] . K[32x64]^T -> two 16x16 f32 tiles ----
        frag16 kf0[2], kf1[2];
#pragma unroll
        for (int c = 0; c < 2; c++) {
            kf0[c].h[0] = *(const v8h*)&Ks[buf][fr][c * 32 + kb];
            kf0[c].h[1] = *(const v8h*)&Ks[buf][fr][c * 32 + kb + 16];
            kf1[c].h[0] = *(const v8h*)&Ks[buf][16 + fr][c * 32 + kb];
            kf1[c].h[1] = *(const v8h*)&Ks[buf][16 + fr][c * 32 + kb + 16];
        }
        v8f s0 = {}, s1 = {};
        s0 = wmma_f16(qf[0].v, kf0[0].v, s0);
        s0 = wmma_f16(qf[1].v, kf0[1].v, s0);
        s1 = wmma_f16(qf[0].v, kf1[0].v, s1);
        s1 = wmma_f16(qf[1].v, kf1[1].v, s1);

        // ---- online softmax (log2 domain, DPP row reductions) ----
        const bool ok0 = (kk + fr) < LKREAL;
        const bool ok1 = (kk + 16 + fr) < LKREAL;
        float p0[8], p1[8], cr[8];
#pragma unroll
        for (int i = 0; i < 8; i++) {
            float a0 = ok0 ? s0[i] * SC : -1e30f;
            float a1 = ok1 ? s1[i] * SC : -1e30f;
            float mx = rowmax16(fmaxf(a0, a1));
            float mn = fmaxf(mrow[i], mx);
            cr[i] = __builtin_exp2f(mrow[i] - mn);
            mrow[i] = mn;
            float e0 = __builtin_exp2f(a0 - mn);
            float e1 = __builtin_exp2f(a1 - mn);
            p0[i] = e0; p1[i] = e1;
            float sm = rowsum16(e0 + e1);
            lrow[i] = lrow[i] * cr[i] + sm;
        }
#pragma unroll
        for (int nc = 0; nc < 4; nc++)
#pragma unroll
            for (int i = 0; i < 8; i++)
                o[nc][i] *= cr[i];

        // ---- P tile (C-layout) -> LDS -> A-layout fragment ----
        const int rbase = (lane >> 4) * 8;
#pragma unroll
        for (int i = 0; i < 8; i++) {
            Pt[wave][rbase + i][fr]      = (_Float16)p0[i];
            Pt[wave][rbase + i][16 + fr] = (_Float16)p1[i];
        }
        asm volatile("s_wait_dscnt 0" ::: "memory");
        frag16 pf;
        pf.h[0] = *(const v8h*)&Pt[wave][fr][kb];
        pf.h[1] = *(const v8h*)&Pt[wave][fr][kb + 16];

        // ---- PV: P[16x32] . V^T[64 x 32keys]^T -> 16x64 accum ----
#pragma unroll
        for (int nc = 0; nc < 4; nc++) {
            frag16 vf;
            vf.h[0] = *(const v8h*)&Vs[buf][nc * 16 + fr][kb];
            vf.h[1] = *(const v8h*)&Vs[buf][nc * 16 + fr][kb + 16];
            o[nc] = wmma_f16(pf.v, vf.v, o[nc]);
        }
        __syncthreads();
        buf ^= 1;
    }

    const int rbase = (lane >> 4) * 8;
#pragma unroll
    for (int i = 0; i < 8; i++) {
        float inv = 1.f / lrow[i];
        int m = q0 + rbase + i;
#pragma unroll
        for (int nc = 0; nc < 4; nc++) {
            int d = h * HEADD + nc * 16 + fr;
            O[((long)(b * LTOK + m)) * CCH + d] = (_Float16)(o[nc][i] * inv);
        }
    }
}

// ---------------- host launch ----------------
extern "C" void kernel_launch(void* const* d_in, const int* in_sizes, int n_in,
                              void* d_out, int out_size, void* d_ws, size_t ws_size,
                              hipStream_t stream) {
    (void)in_sizes; (void)n_in; (void)out_size; (void)ws_size;
    const float* x0    = (const float*)d_in[0];
    const float* cond  = (const float*)d_in[1];
    const float* gamma = (const float*)d_in[2];
    const float* beta  = (const float*)d_in[3];
    const float* qkv_w = (const float*)d_in[4];
    const float* qkv_b = (const float*)d_in[5];
    const float* ckv_w = (const float*)d_in[6];
    const float* ckv_b = (const float*)d_in[7];
    const float* out_w = (const float*)d_in[8];
    const float* out_b = (const float*)d_in[9];
    float* out = (float*)d_out;

    char* ws = (char*)d_ws;
    size_t off = 0;
    auto alloc = [&](size_t bytes) -> void* {
        void* p = ws + off;
        off += (bytes + 255) & ~(size_t)255;
        return p;
    };
    float*    stats  = (float*)alloc((size_t)BATCH * NGROUPS * 2 * sizeof(float));
    _Float16* x16    = (_Float16*)alloc((size_t)BATCH * LTOK * CCH * 2);       // reused for attn out
    _Float16* qkv16  = (_Float16*)alloc((size_t)BATCH * LTOK * 3 * CCH * 2);
    _Float16* cond16 = (_Float16*)alloc((size_t)BATCH * LCP * CONDC * 2);
    _Float16* kvc16  = (_Float16*)alloc((size_t)BATCH * LCP * 2 * CCH * 2);
    _Float16* wq16   = (_Float16*)alloc((size_t)3 * CCH * CCH * 2);
    _Float16* wkv16  = (_Float16*)alloc((size_t)2 * CCH * CONDC * 2);
    _Float16* wout16 = (_Float16*)alloc((size_t)CCH * CCH * 2);
    _Float16* Kh     = (_Float16*)alloc((size_t)BATCH * NHEADS * LKP * HEADD * 2);
    _Float16* Vt     = (_Float16*)alloc((size_t)BATCH * NHEADS * LKP * HEADD * 2);

    // 1) GroupNorm
    gn_stats_kernel<<<BATCH * NGROUPS, 256, 0, stream>>>(x0, stats);
    gn_apply_kernel<<<BATCH * CCH, 256, 0, stream>>>(x0, stats, gamma, beta, x16);

    // 2) weight / cond conversion
    cvt_f16_kernel<<<(3 * CCH * CCH) / 256, 256, 0, stream>>>(qkv_w, wq16, 3 * CCH * CCH);
    cvt_f16_kernel<<<(2 * CCH * CONDC) / 256, 256, 0, stream>>>(ckv_w, wkv16, 2 * CCH * CONDC);
    cvt_f16_kernel<<<(CCH * CCH) / 256, 256, 0, stream>>>(out_w, wout16, CCH * CCH);
    pad_cond_kernel<<<(BATCH * LCP * CONDC) / 256, 256, 0, stream>>>(cond, cond16);

    // 3) QKV GEMM: [1024,512] x [1536,512]^T per batch
    {
        dim3 grid((LTOK / BM) * (3 * CCH / BN), BATCH);
        gemm_kernel<0><<<grid, 128, 0, stream>>>(
            x16, (long)LTOK * CCH, CCH, wq16, CCH, qkv_b,
            qkv16, (long)LTOK * 3 * CCH, 3 * CCH,
            nullptr, nullptr, 0, LTOK, 3 * CCH, CCH);
    }
    // 4) cond KV GEMM: [128,768] x [1024,768]^T per batch
    {
        dim3 grid((LCP / BM) * (2 * CCH / BN), BATCH);
        gemm_kernel<0><<<grid, 128, 0, stream>>>(
            cond16, (long)LCP * CONDC, CONDC, wkv16, CONDC, ckv_b,
            kvc16, (long)LCP * 2 * CCH, 2 * CCH,
            nullptr, nullptr, 0, LCP, 2 * CCH, CONDC);
    }
    // 5) repack K / V^T (padded to 1120 keys)
    {
        long total = (long)BATCH * NHEADS * LKP * HEADD;
        build_kv_kernel<<<(int)((total + 255) / 256), 256, 0, stream>>>(qkv16, kvc16, Kh, Vt);
    }
    // 6) attention -> x16 (reused as attention output buffer)
    attn_kernel<<<BATCH * NHEADS * (LTOK / 64), 128, 0, stream>>>(qkv16, Kh, Vt, x16);

    // 7) output projection + bias + transpose back + residual -> d_out
    {
        dim3 grid((LTOK / BM) * (CCH / BN), BATCH);
        gemm_kernel<1><<<grid, 128, 0, stream>>>(
            x16, (long)LTOK * CCH, CCH, wout16, CCH, out_b,
            nullptr, 0, 0,
            out, x0, (long)CCH * LTOK, LTOK, CCH, CCH);
    }
}
